// AttentionLayers_85289460564565
// MI455X (gfx1250) — compile-verified, matrix operand order checked by reading, not run on hardware
//
#include <hip/hip_runtime.h>
#include <hip/hip_bf16.h>

// ---------------------------------------------------------------------------
// Problem constants (from reference): B=2, N=2048, DIM=1024, H=16, DH=64,
// MEM=16, ROT=32.  bf16 WMMA (16x16x32, f32 acc) for all four GEMM stages
// AND for the pre/post head-mixing (K=16 padded to 32); only the masked
// online softmax scan stays on VALU.
// ---------------------------------------------------------------------------
#define BDIM   2
#define NSEQ   2048
#define DMODEL 1024
#define NH     16
#define DH     64
#define MEMN   16
#define NPAD   2080            // 16 mem + 2048 seq + 16 zero pad (mult of 32)

typedef __bf16 v16bf __attribute__((ext_vector_type(16)));
typedef float  v8f   __attribute__((ext_vector_type(8)));
typedef float  f4v   __attribute__((ext_vector_type(4)));

static __device__ __forceinline__ unsigned short f2bf(float f) {
  return __builtin_bit_cast(unsigned short, (__bf16)f);
}
static __device__ __forceinline__ float bf2f(unsigned short u) {
  unsigned int x = ((unsigned int)u) << 16;
  return __builtin_bit_cast(float, x);
}
static __device__ __forceinline__ v8f v8f_zero() {
  v8f z;
#pragma unroll
  for (int e = 0; e < 8; ++e) z[e] = 0.f;
  return z;
}

union FragU { v16bf v; f4v f[2]; };

// 16x32 bf16 A-style fragment: lane supplies row (lane&15); K halves at
// k0 + (lane>>4)*8 + {0..7} and +16 + {0..7}  (CDNA5 16-bit A striping).
// B operands are loaded as A-fragments of B^T (symmetric striping assumed).
static __device__ __forceinline__ v16bf
load_frag(const unsigned short* base, int row0, int ld, int k0, int lane) {
  const unsigned short* p =
      base + (size_t)(row0 + (lane & 15)) * ld + k0 + ((lane >> 4) << 3);
  FragU u;
  u.f[0] = *(const f4v*)p;
  u.f[1] = *(const f4v*)(p + 16);
  return u.v;
}

static __device__ __forceinline__ v8f wmma_bf16(v16bf a, v16bf b, v8f c) {
  return __builtin_amdgcn_wmma_f32_16x16x32_bf16(false, a, false, b,
                                                 (short)0, c, false, false);
}

// ---------------------------------------------------------------------------
// f32 -> bf16 conversion (grid-stride)
// ---------------------------------------------------------------------------
__global__ void cvt_kernel(const float* __restrict__ in,
                           unsigned short* __restrict__ out, int n) {
  for (int i = blockIdx.x * blockDim.x + threadIdx.x; i < n;
       i += gridDim.x * blockDim.x)
    out[i] = f2bf(in[i]);
}

// ---------------------------------------------------------------------------
// mem_k/mem_v -> rows [0,16) of Kf/Vt, zero the pad rows [2064,2080)
//   Kf layout: ((b*NH+h)*NPAD + j)*DH + d          (K, row-major j,d)
//   Vt layout: ((b*NH+h)*DH + d)*NPAD + j          (V^T, row-major d,j)
// ---------------------------------------------------------------------------
__global__ void prep_mem_kernel(const float* __restrict__ mk,
                                const float* __restrict__ mv,
                                unsigned short* __restrict__ Kf,
                                unsigned short* __restrict__ Vt) {
  int idx = blockIdx.x * blockDim.x + threadIdx.x;   // < 2*16*32*64
  int d  = idx & 63;
  int j2 = (idx >> 6) & 31;
  int h  = (idx >> 11) & 15;
  int b  = idx >> 15;
  if (b >= BDIM) return;
  int bh = b * NH + h;
  unsigned short kv = 0, vv = 0;
  int j;
  if (j2 < MEMN) {
    j  = j2;
    kv = f2bf(mk[(h * MEMN + j2) * DH + d]);
    vv = f2bf(mv[(h * MEMN + j2) * DH + d]);
  } else {
    j = MEMN + NSEQ + (j2 - MEMN);                   // pad rows -> zero
  }
  Kf[((size_t)bh * NPAD + j) * DH + d]  = kv;
  Vt[((size_t)bh * DH + d) * NPAD + j]  = vv;
}

// ---------------------------------------------------------------------------
// Rotary on Q (layout (b,n,h*64+d)) and on Kf (layout (b,h,j,d), j=16+n)
// rot[n*32+p]: first 16 entries == last 16, so one angle per pair.
// ---------------------------------------------------------------------------
__global__ void rotary_q_kernel(const float* __restrict__ rot,
                                unsigned short* __restrict__ Q) {
  int idx = blockIdx.x * blockDim.x + threadIdx.x;   // (b,n,h,p), p<16
  if (idx >= BDIM * NSEQ * NH * 16) return;
  int p = idx & 15, h = (idx >> 4) & 15, n = (idx >> 8) & 2047, b = idx >> 19;
  float a = rot[n * 32 + p];
  float c = __cosf(a), s = __sinf(a);
  size_t base = ((size_t)(b * NSEQ + n)) * DMODEL + h * DH;
  float x0 = bf2f(Q[base + p]), x1 = bf2f(Q[base + p + 16]);
  Q[base + p]      = f2bf(x0 * c - x1 * s);
  Q[base + p + 16] = f2bf(x1 * c + x0 * s);
}

__global__ void rotary_k_kernel(const float* __restrict__ rot,
                                unsigned short* __restrict__ Kf) {
  int idx = blockIdx.x * blockDim.x + threadIdx.x;
  if (idx >= BDIM * NSEQ * NH * 16) return;
  int p = idx & 15, h = (idx >> 4) & 15, n = (idx >> 8) & 2047, b = idx >> 19;
  float a = rot[n * 32 + p];
  float c = __cosf(a), s = __sinf(a);
  size_t base = ((size_t)(b * NH + h) * NPAD + MEMN + n) * DH;
  float x0 = bf2f(Kf[base + p]), x1 = bf2f(Kf[base + p + 16]);
  Kf[base + p]      = f2bf(x0 * c - x1 * s);
  Kf[base + p + 16] = f2bf(x1 * c + x0 * s);
}

// ---------------------------------------------------------------------------
// Tiled bf16 WMMA GEMM:  C(4096x1024) = A(4096x1024) @ W^T(1024x1024)
// 4 waves / block, each wave owns a 32x64 C tile (2x4 WMMA tiles),
// K loop in steps of 32 -> 8 v_wmma per step.  Streams are L2-resident.
// Epilogue modes: 0=Q plain bf16, 1=scatter into Kf, 2=scatter into Vt,
// 3=f32 out + bias (final projection).
// ---------------------------------------------------------------------------
__global__ __launch_bounds__(128) void
gemm_kernel(const unsigned short* __restrict__ A,
            const unsigned short* __restrict__ W,
            const float* __restrict__ bias,
            unsigned short* __restrict__ outB,
            float* __restrict__ outF, int mode) {
  const int lane = threadIdx.x & 31;
  const int wave = threadIdx.x >> 5;
  const int tile = blockIdx.x * 4 + wave;            // 2048 wave tiles
  const int row0 = (tile >> 4) * 32;                 // 128 row tiles
  const int col0 = (tile & 15) * 64;                 // 16 col tiles

  v8f acc[2][4];
#pragma unroll
  for (int ri = 0; ri < 2; ++ri)
#pragma unroll
    for (int ci = 0; ci < 4; ++ci) acc[ri][ci] = v8f_zero();

  for (int k0 = 0; k0 < DMODEL; k0 += 32) {
    v16bf af[2], bfr[4];
#pragma unroll
    for (int ri = 0; ri < 2; ++ri)
      af[ri] = load_frag(A, row0 + ri * 16, DMODEL, k0, lane);
#pragma unroll
    for (int ci = 0; ci < 4; ++ci)
      bfr[ci] = load_frag(W, col0 + ci * 16, DMODEL, k0, lane);
#pragma unroll
    for (int ri = 0; ri < 2; ++ri)
#pragma unroll
      for (int ci = 0; ci < 4; ++ci)
        acc[ri][ci] = wmma_bf16(af[ri], bfr[ci], acc[ri][ci]);
  }

#pragma unroll
  for (int ri = 0; ri < 2; ++ri) {
#pragma unroll
    for (int ci = 0; ci < 4; ++ci) {
#pragma unroll
      for (int e = 0; e < 8; ++e) {
        int r = row0 + ri * 16 + e + ((lane >> 4) << 3);   // C-tile M map
        int c = col0 + ci * 16 + (lane & 15);              // C-tile N map
        float val = acc[ri][ci][e];
        if (mode == 3) {
          outF[(size_t)r * DMODEL + c] = val + bias[c];
        } else if (mode == 0) {
          outB[(size_t)r * DMODEL + c] = f2bf(val);
        } else {
          int b = r >> 11, n = r & 2047, h = c >> 6, d = c & 63;
          int bh = b * NH + h;
          if (mode == 1)
            outB[((size_t)bh * NPAD + MEMN + n) * DH + d] = f2bf(val);
          else
            outB[((size_t)bh * DH + d) * NPAD + MEMN + n] = f2bf(val);
        }
      }
    }
  }
}

// ---------------------------------------------------------------------------
// Fused attention.  One block = (batch b, 16 query rows), all 16 heads,
// 8 waves.  j-tiles of 32.  Head-mixings run on WMMA:
//   buf1 = dots^T   [i][j][h(pad32)] bf16   (later aliased as p^T [i][j][k])
//   buf2 = mixed    [k][i][j]        bf16   (logits, later attn2 probs)
// pass1: QK^T(wmma) -> buf1 -> premix(wmma) -> buf2 -> online max/sum
// pass2: QK^T(wmma) -> premix(wmma) -> exp/l -> postmix(wmma) -> attn@V(wmma)
// ---------------------------------------------------------------------------
static __device__ __forceinline__ void
compute_dots_tile(const unsigned short* __restrict__ Kf,
                  const v16bf (&qf)[2][2], unsigned short* sdt,
                  int b, int wave, int lane, int j0, float scale) {
#pragma unroll
  for (int hl = 0; hl < 2; ++hl) {
    int h = wave * 2 + hl;
    const unsigned short* kb = Kf + ((size_t)(b * NH + h)) * NPAD * DH;
#pragma unroll
    for (int jc = 0; jc < 2; ++jc) {
      v8f d = v8f_zero();
      v16bf b0 = load_frag(kb, j0 + jc * 16, DH, 0, lane);
      v16bf b1 = load_frag(kb, j0 + jc * 16, DH, 32, lane);
      d = wmma_bf16(qf[hl][0], b0, d);
      d = wmma_bf16(qf[hl][1], b1, d);
#pragma unroll
      for (int e = 0; e < 8; ++e) {
        int i = e + ((lane >> 4) << 3);
        int j = jc * 16 + (lane & 15);
        sdt[i * 1024 + j * 32 + h] = f2bf(d[e] * scale);   // [i][j][h]
      }
    }
  }
}

// D(k x 16j) = M(16k x 32pad) @ src_i(32pad x 16j), per i owned by this wave.
// src rows [i][j][0..31] are contiguous -> B loaded as A-frag of B^T.
static __device__ __forceinline__ void
mix_tile(const unsigned short* src, unsigned short* dst, v16bf mfrag,
         int wave, int lane) {
#pragma unroll
  for (int il = 0; il < 2; ++il) {
    int i = wave * 2 + il;
#pragma unroll
    for (int jc = 0; jc < 2; ++jc) {
      v8f d2 = v8f_zero();
      v16bf bfrag = load_frag(src + i * 1024, jc * 16, 32, 0, lane);
      d2 = wmma_bf16(mfrag, bfrag, d2);
#pragma unroll
      for (int e = 0; e < 8; ++e) {
        int k = e + ((lane >> 4) << 3);
        int j = jc * 16 + (lane & 15);
        dst[(k * 16 + i) * 32 + j] = f2bf(d2[e]);          // [k][i][j]
      }
    }
  }
}

__global__ __launch_bounds__(256) void
attention_kernel(const unsigned short* __restrict__ Q,
                 const unsigned short* __restrict__ Kf,
                 const unsigned short* __restrict__ Vt,
                 const float* __restrict__ pre,
                 const float* __restrict__ post,
                 unsigned short* __restrict__ AO) {
  __shared__ unsigned short s_buf1[16 * 32 * 32];   // 32 KB: dots^T / p^T
  __shared__ unsigned short s_buf2[16 * 16 * 32];   // 16 KB: mixed [k][i][j]
  __shared__ unsigned short s_preT[16 * 32];        // pre^T,  K padded to 32
  __shared__ unsigned short s_postT[16 * 32];       // post^T, K padded to 32

  const int tid  = threadIdx.x;
  const int lane = tid & 31, wave = tid >> 5;
  const int b  = blockIdx.y;
  const int i0 = blockIdx.x * 16;
  const int kk = tid >> 4, ii = tid & 15;       // softmax row owned by thread
  const float scale = 0.125f;                   // 64^-0.5

  // pre^T[k][h] = pre[h][k], post^T[k2][k] = post[k][k2]; upper K half = 0
#pragma unroll
  for (int t = tid; t < 512; t += 256) {
    int r = t >> 5, c = t & 31;
    s_preT[t]  = (c < NH) ? f2bf(pre[c * NH + r])  : (unsigned short)0;
    s_postT[t] = (c < NH) ? f2bf(post[c * NH + r]) : (unsigned short)0;
  }
  // zero buf1's padded K half [i][j][16..31] once; never written again
  for (int t = tid; t < 16 * 32 * 16; t += 256) {
    int h = 16 + (t & 15), j = (t >> 4) & 31, i = t >> 9;
    s_buf1[i * 1024 + j * 32 + h] = 0;
  }

  // Q fragments for this wave's two heads, reused across the whole j loop
  v16bf qf[2][2];
#pragma unroll
  for (int hl = 0; hl < 2; ++hl) {
    int h = wave * 2 + hl;
    const unsigned short* qb = Q + ((size_t)(b * NSEQ + i0)) * DMODEL + h * DH;
    qf[hl][0] = load_frag(qb, 0, DMODEL, 0, lane);
    qf[hl][1] = load_frag(qb, 0, DMODEL, 32, lane);
  }
  __syncthreads();
  v16bf prefrag  = load_frag(s_preT, 0, 32, 0, lane);
  v16bf postfrag = load_frag(s_postT, 0, 32, 0, lane);

  const int njt = (i0 + 63) / 32;               // j-tiles covering j<=i0+31
  float m_run = -3.402823466e38f, l_run = 0.f;

  // ------------------------------ pass 1: softmax statistics
  for (int jt = 0; jt < njt; ++jt) {
    int j0 = jt * 32;
    compute_dots_tile(Kf, qf, s_buf1, b, wave, lane, j0, scale);
    __syncthreads();
    mix_tile(s_buf1, s_buf2, prefrag, wave, lane);      // pre_proj via WMMA
    __syncthreads();
    for (int jj = 0; jj < 32; ++jj) {
      float d2 = bf2f(s_buf2[(kk * 16 + ii) * 32 + jj]);
      int jg = j0 + jj;
      if (!((jg - MEMN > i0 + ii) || (jg >= MEMN + NSEQ))) {
        float nm = fmaxf(m_run, d2);
        l_run = l_run * __expf(m_run - nm) + __expf(d2 - nm);
        m_run = nm;
      }
    }
    __syncthreads();
  }
  const float inv_l = 1.f / l_run;

  // ------------------------------ pass 2: probs, post-mix, attn@V
  v8f oacc[2][4];
#pragma unroll
  for (int hl = 0; hl < 2; ++hl)
#pragma unroll
    for (int dt = 0; dt < 4; ++dt) oacc[hl][dt] = v8f_zero();

  for (int jt = 0; jt < njt; ++jt) {
    int j0 = jt * 32;
    compute_dots_tile(Kf, qf, s_buf1, b, wave, lane, j0, scale);
    __syncthreads();
    mix_tile(s_buf1, s_buf2, prefrag, wave, lane);      // pre_proj via WMMA
    __syncthreads();
    for (int jj = 0; jj < 32; ++jj) {                   // exp / l, masked
      float d2 = bf2f(s_buf2[(kk * 16 + ii) * 32 + jj]);
      int jg = j0 + jj;
      bool masked = (jg - MEMN > i0 + ii) || (jg >= MEMN + NSEQ);
      float p = masked ? 0.f : __expf(d2 - m_run) * inv_l;
      s_buf1[ii * 1024 + jj * 32 + kk] = f2bf(p);       // p^T [i][j][k]
    }
    __syncthreads();
    mix_tile(s_buf1, s_buf2, postfrag, wave, lane);     // post_proj via WMMA
    __syncthreads();
#pragma unroll
    for (int hl = 0; hl < 2; ++hl) {                    // attn2 @ V (K = 32 j)
      int k2 = wave * 2 + hl;
      const unsigned short* arow = s_buf2 + (size_t)k2 * 16 * 32;
      v16bf afrag = load_frag(arow, 0, 32, 0, lane);
      const unsigned short* vb = Vt + ((size_t)(b * NH + k2)) * DH * NPAD;
#pragma unroll
      for (int dt = 0; dt < 4; ++dt) {
        v16bf vf = load_frag(vb, dt * 16, NPAD, j0, lane);
        oacc[hl][dt] = wmma_bf16(afrag, vf, oacc[hl][dt]);
      }
    }
    __syncthreads();
  }

  // write (b, n, k2*64+d) bf16
#pragma unroll
  for (int hl = 0; hl < 2; ++hl) {
#pragma unroll
    for (int dt = 0; dt < 4; ++dt) {
#pragma unroll
      for (int e = 0; e < 8; ++e) {
        int n = i0 + e + ((lane >> 4) << 3);
        int c = (wave * 2 + hl) * DH + dt * 16 + (lane & 15);
        AO[((size_t)(b * NSEQ + n)) * DMODEL + c] = f2bf(oacc[hl][dt][e]);
      }
    }
  }
}

// ---------------------------------------------------------------------------
extern "C" void kernel_launch(void* const* d_in, const int* in_sizes, int n_in,
                              void* d_out, int out_size, void* d_ws,
                              size_t ws_size, hipStream_t stream) {
  const float* x    = (const float*)d_in[0];
  const float* rot  = (const float*)d_in[1];
  const float* Wq   = (const float*)d_in[2];
  const float* Wk   = (const float*)d_in[3];
  const float* Wv   = (const float*)d_in[4];
  const float* mk   = (const float*)d_in[5];
  const float* mv   = (const float*)d_in[6];
  const float* pre  = (const float*)d_in[7];
  const float* post = (const float*)d_in[8];
  const float* Wo   = (const float*)d_in[9];
  const float* bo   = (const float*)d_in[10];
  float* out = (float*)d_out;

  char* ws = (char*)d_ws;
  size_t off = 0;
  auto alloc = [&](size_t bytes) {
    size_t o = off;
    off = (off + bytes + 255) & ~(size_t)255;
    return o;
  };
  const size_t nX = (size_t)BDIM * NSEQ * DMODEL;          // 4M elems
  const size_t nW = (size_t)DMODEL * DMODEL;               // 1M elems
  const size_t nKV = (size_t)BDIM * NH * NPAD * DH;        // padded K/V

  unsigned short* xb  = (unsigned short*)(ws + alloc(nX * 2));
  unsigned short* Wqb = (unsigned short*)(ws + alloc(nW * 2));
  unsigned short* Wkb = (unsigned short*)(ws + alloc(nW * 2));
  unsigned short* Wvb = (unsigned short*)(ws + alloc(nW * 2));
  unsigned short* Wob = (unsigned short*)(ws + alloc(nW * 2));
  unsigned short* Qb  = (unsigned short*)(ws + alloc(nX * 2));
  unsigned short* Kf  = (unsigned short*)(ws + alloc(nKV * 2));
  unsigned short* Vt  = (unsigned short*)(ws + alloc(nKV * 2));
  unsigned short* AOb = (unsigned short*)(ws + alloc(nX * 2));

  // 1) bf16 conversions
  cvt_kernel<<<4096, 256, 0, stream>>>(x,  xb,  (int)nX);
  cvt_kernel<<<1024, 256, 0, stream>>>(Wq, Wqb, (int)nW);
  cvt_kernel<<<1024, 256, 0, stream>>>(Wk, Wkb, (int)nW);
  cvt_kernel<<<1024, 256, 0, stream>>>(Wv, Wvb, (int)nW);
  cvt_kernel<<<1024, 256, 0, stream>>>(Wo, Wob, (int)nW);

  // 2) memory slots + zero padding of Kf/Vt
  prep_mem_kernel<<<256, 256, 0, stream>>>(mk, mv, Kf, Vt);

  // 3) QKV projections (WMMA GEMMs); K/V scattered into attention layouts
  gemm_kernel<<<512, 128, 0, stream>>>(xb, Wqb, nullptr, Qb, nullptr, 0);
  gemm_kernel<<<512, 128, 0, stream>>>(xb, Wkb, nullptr, Kf, nullptr, 1);
  gemm_kernel<<<512, 128, 0, stream>>>(xb, Wvb, nullptr, Vt, nullptr, 2);

  // 4) rotary embedding on Q and K (seq rows only)
  rotary_q_kernel<<<4096, 256, 0, stream>>>(rot, Qb);
  rotary_k_kernel<<<4096, 256, 0, stream>>>(rot, Kf);

  // 5) fused attention: WMMA dots, WMMA head-mixing, WMMA attn@V
  attention_kernel<<<dim3(NSEQ / 16, BDIM), 256, 0, stream>>>(Qb, Kf, Vt, pre,
                                                              post, AOb);

  // 6) output projection + bias (WMMA GEMM, f32 out)
  gemm_kernel<<<512, 128, 0, stream>>>(AOb, Wob, bo, nullptr, out, 3);
}
